// IterativeWTA_35115652612493
// MI455X (gfx1250) — 8-bit matrix operand order repaired and checked by reading, compiled, not run on hardware
//
#include <hip/hip_runtime.h>
#include <hip/hip_bf16.h>
#include <stdint.h>

typedef __attribute__((ext_vector_type(8))) int v8i;

#define N_ROWS 4096
#define NX 1024
#define NH 2048
#define NY 2048
#define NHY 4096
#define MAX_ITERS 32

#define OFF_X8   ((size_t)0)
#define OFF_WXP  (OFF_X8   + (size_t)N_ROWS*NX)
#define OFF_BHP  (OFF_WXP  + (size_t)NX*NHY)
#define OFF_BYP  (OFF_BHP  + (size_t)NHY*NH)
#define OFF_H0Y0 (OFF_BYP  + (size_t)NHY*NY)
#define OFF_HY0  (OFF_H0Y0 + (size_t)N_ROWS*NHY*2)
#define OFF_HY1  (OFF_HY0  + (size_t)N_ROWS*NHY)
#define OFF_TST  (OFF_HY1  + (size_t)N_ROWS*NHY)

__global__ void pack_binary_i8(const float* __restrict__ src,
                               signed char* __restrict__ dst, int n) {
  int i = blockIdx.x * blockDim.x + threadIdx.x;
  if (i < n) dst[i] = (signed char)(src[i] > 0.5f);
}

template <int CMODE>
__global__ void pack_B_frag(const float* __restrict__ a, const float* __restrict__ b,
                            signed char* __restrict__ dst, int K, int N) {
  const int ktiles = K >> 6;
  const size_t total = (size_t)(N >> 4) * ktiles * 1024;
  for (size_t i = (size_t)blockIdx.x * blockDim.x + threadIdx.x; i < total;
       i += (size_t)gridDim.x * blockDim.x) {
    int j    = (int)(i & 31);
    int lane = (int)((i >> 5) & 31);
    size_t tile = i >> 10;
    int nt = (int)(tile / ktiles), kt = (int)(tile % ktiles);
    int n = nt * 16 + (lane & 15);
    int k = kt * 64 + ((lane >> 4) << 4) + j + ((j >> 4) << 4);
    int bit, neg = 0;
    if (CMODE == 0) {
      int half = N >> 1;
      float v = (n < half) ? a[(size_t)k * half + n] : b[(size_t)k * half + (n - half)];
      bit = v > 0.5f;
    } else {
      int half = K >> 1;
      if (k < half) { bit = a[(size_t)k * N + n] > 0.5f; neg = 1; }
      else          { bit = b[(size_t)(k - half) * N + n] > 0.5f; }
    }
    dst[i] = (signed char)(neg ? -bit : bit);
  }
}

template <int MODE>
__global__ __launch_bounds__(128)
void gemm_iu8(const signed char* __restrict__ A, int lda,
              const signed char* __restrict__ PB0, const signed char* __restrict__ PB1,
              int K, short* __restrict__ bias_out,
              signed char* __restrict__ HYnext,
              const int* __restrict__ tstart, int iter) {
  __shared__ __align__(16) signed char As[2][64 * 80];
  __shared__ __align__(16) signed char Bls[2][8192];
  const int tid  = threadIdx.x;
  const int wave = tid >> 5;
  const int lane = tid & 31;
  const int n0   = blockIdx.x * 128;
  const int m0   = blockIdx.y * 64;
  const int zsel = (MODE == 1) ? (int)blockIdx.z : 0;
  const signed char* PB = (MODE == 1 && zsel == 1) ? PB1 : PB0;
  const int ktiles = K >> 6;

  int t = 0;
  if (MODE == 1) {
    t = tstart[0] - iter;
    if (t < 1) {
      for (int i = tid; i < 512; i += 128) {
        int row = i >> 3, c = (i & 7) * 16;
        size_t idx = (size_t)(m0 + row) * NHY + (size_t)zsel * NH + n0 + c;
        *(int4*)(HYnext + idx) = *(const int4*)(A + idx);
      }
      return;
    }
  }

  v8i acc[8] = {};
  const int arow = wave * 16 + (lane & 15);
  const int akb  = (lane >> 4) << 4;
  const int ldrow = tid >> 1;
  const int ldcb  = (tid & 1) * 32;
  const signed char* arowp = A + (size_t)(m0 + ldrow) * lda + ldcb;
  const int btn  = tid >> 4;
  const int boff = (tid & 15) * 64;
  const signed char* bsrc = PB + ((size_t)((n0 >> 4) + btn) * ktiles) * 1024 + boff;

  {
    const int4* sa = (const int4*)(arowp);
    int4 a0 = sa[0], a1 = sa[1];
    const int4* sb = (const int4*)(bsrc);
    int4 b0 = sb[0], b1 = sb[1], b2 = sb[2], b3 = sb[3];
    int4* da = (int4*)(&As[0][0] + ldrow * 80 + ldcb);
    da[0] = a0; da[1] = a1;
    int4* db = (int4*)(&Bls[0][0] + btn * 1024 + boff);
    db[0] = b0; db[1] = b1; db[2] = b2; db[3] = b3;
  }
  __syncthreads();

  for (int kt = 0; kt < ktiles; ++kt) {
    const bool more = (kt + 1 < ktiles);
    int4 na0, na1, nb0, nb1, nb2, nb3;
    if (more) {
      const int4* sa = (const int4*)(arowp + (kt + 1) * 64);
      na0 = sa[0]; na1 = sa[1];
      const int4* sb = (const int4*)(bsrc + (kt + 1) * 1024);
      nb0 = sb[0]; nb1 = sb[1]; nb2 = sb[2]; nb3 = sb[3];
    }

    v8i a;
    {
      const signed char* Ab = &As[kt & 1][0] + arow * 80 + akb;
      int4 p0 = *(const int4*)(Ab +  0); a[0] = p0.x; a[1] = p0.y; a[2] = p0.z; a[3] = p0.w;
      int4 p1 = *(const int4*)(Ab + 32); a[4] = p1.x; a[5] = p1.y; a[6] = p1.z; a[7] = p1.w;
    }
    const signed char* Bb = &Bls[kt & 1][0] + lane * 32;
#pragma unroll
    for (int tn = 0; tn < 8; ++tn) {
      v8i bf = *(const v8i*)(Bb + tn * 1024);
      acc[tn] = __builtin_amdgcn_wmma_i32_16x16x64_iu8(true, a, true, bf, acc[tn],
                                                       false, false);
    }

    if (more) {
      int4* da = (int4*)(&As[(kt + 1) & 1][0] + ldrow * 80 + ldcb);
      da[0] = na0; da[1] = na1;
      int4* db = (int4*)(&Bls[(kt + 1) & 1][0] + btn * 1024 + boff);
      db[0] = nb0; db[1] = nb1; db[2] = nb2; db[3] = nb3;
    }
    __syncthreads();
  }

  const int cn  = lane & 15;
  const int cmb = (lane >> 4) << 3;
#pragma unroll
  for (int tn = 0; tn < 8; ++tn) {
    int n = n0 + tn * 16 + cn;
#pragma unroll
    for (int r = 0; r < 8; ++r) {
      int m = m0 + wave * 16 + cmb + r;
      int v = acc[tn][r];
      if (MODE == 0) {
        bias_out[(size_t)m * NHY + n] = (short)v;
      } else {
        size_t idx = (size_t)m * NHY + (size_t)zsel * NH + n;
        int z = ((int)bias_out[idx] + v) >= t;
        HYnext[idx] = (signed char)((int)A[idx] | z);
      }
    }
  }
}

__global__ void reduce_max_i16(const short* __restrict__ d, int n, int* __restrict__ out) {
  __shared__ int s[256];
  int v = 0;
  for (int i = blockIdx.x * blockDim.x + threadIdx.x; i < n; i += gridDim.x * blockDim.x)
    v = max(v, (int)d[i]);
  s[threadIdx.x] = v;
  __syncthreads();
  for (int st = 128; st > 0; st >>= 1) {
    if (threadIdx.x < st) s[threadIdx.x] = max(s[threadIdx.x], s[threadIdx.x + st]);
    __syncthreads();
  }
  if (threadIdx.x == 0) atomicMax(out, s[0]);
}

__global__ __launch_bounds__(256)
void finalize(const signed char* __restrict__ HY, const short* __restrict__ H0Y0,
              const float* __restrict__ why, float* __restrict__ out) {
  const int row = blockIdx.x;
  const int tid = threadIdx.x;
  __shared__ int s_any;
  __shared__ int s_val[256];
  __shared__ int s_idx[256];
  if (tid == 0) s_any = 0;
  __syncthreads();
  const signed char* hrow = HY + (size_t)row * NHY;
  const signed char* yrow = hrow + NH;
  int any = 0;
  for (int j = tid; j < NY; j += 256) any |= yrow[j];
  if (any) s_any = 1;
  __syncthreads();
  float* oh = out + (size_t)row * NH;
  float* oy = out + (size_t)N_ROWS * NH + (size_t)row * NY;
  if (s_any) {
    for (int j = tid; j < NH; j += 256) oh[j] = (float)hrow[j];
    for (int j = tid; j < NY; j += 256) oy[j] = (float)yrow[j];
    return;
  }
  const short* h0r = H0Y0 + (size_t)row * NHY;
  const short* y0r = h0r + NH;
  int bv = -(1 << 30), bi = 0;
  for (int j = tid; j < NH; j += 256) { int v = h0r[j]; if (v > bv) { bv = v; bi = j; } }
  s_val[tid] = bv; s_idx[tid] = bi;
  __syncthreads();
  for (int st = 128; st > 0; st >>= 1) {
    if (tid < st) {
      if (s_val[tid + st] > s_val[tid] ||
          (s_val[tid + st] == s_val[tid] && s_idx[tid + st] < s_idx[tid])) {
        s_val[tid] = s_val[tid + st]; s_idx[tid] = s_idx[tid + st];
      }
    }
    __syncthreads();
  }
  const int hidx = s_idx[0];
  __syncthreads();
  bv = -(1 << 30); bi = 0;
  for (int j = tid; j < NY; j += 256) {
    int v = (int)y0r[j] - (int)(why[(size_t)hidx * NY + j] > 0.5f);
    if (v > bv) { bv = v; bi = j; }
  }
  s_val[tid] = bv; s_idx[tid] = bi;
  __syncthreads();
  for (int st = 128; st > 0; st >>= 1) {
    if (tid < st) {
      if (s_val[tid + st] > s_val[tid] ||
          (s_val[tid + st] == s_val[tid] && s_idx[tid + st] < s_idx[tid])) {
        s_val[tid] = s_val[tid + st]; s_idx[tid] = s_idx[tid + st];
      }
    }
    __syncthreads();
  }
  const int yidx = s_idx[0];
  for (int j = tid; j < NH; j += 256) oh[j] = (j == hidx) ? 1.0f : 0.0f;
  for (int j = tid; j < NY; j += 256) oy[j] = (j == yidx) ? 1.0f : 0.0f;
}

extern "C" void kernel_launch(void* const* d_in, const int* in_sizes, int n_in,
                              void* d_out, int out_size, void* d_ws, size_t ws_size,
                              hipStream_t stream) {
  (void)in_sizes; (void)n_in; (void)out_size; (void)ws_size;
  const float* x   = (const float*)d_in[0];
  const float* wxy = (const float*)d_in[1];
  const float* wxh = (const float*)d_in[2];
  const float* why = (const float*)d_in[3];
  const float* wyy = (const float*)d_in[4];
  const float* whh = (const float*)d_in[5];
  const float* wyh = (const float*)d_in[6];
  float* out = (float*)d_out;
  char*  ws  = (char*)d_ws;

  signed char* X8  = (signed char*)(ws + OFF_X8);
  signed char* WXP = (signed char*)(ws + OFF_WXP);
  signed char* BHP = (signed char*)(ws + OFF_BHP);
  signed char* BYP = (signed char*)(ws + OFF_BYP);
  short*       H0  = (short*)(ws + OFF_H0Y0);
  signed char* HY0 = (signed char*)(ws + OFF_HY0);
  signed char* HY1 = (signed char*)(ws + OFF_HY1);
  int*         TST = (int*)(ws + OFF_TST);

  hipMemsetAsync(HY0, 0, (size_t)N_ROWS * NHY, stream);
  hipMemsetAsync(TST, 0, sizeof(int), stream);

  { int n = N_ROWS * NX; pack_binary_i8<<<(n + 255) / 256, 256, 0, stream>>>(x, X8, n); }
  pack_B_frag<0><<<4096, 256, 0, stream>>>(wxh, wxy, WXP, NX, NHY);
  pack_B_frag<1><<<4096, 256, 0, stream>>>(whh, wyh, BHP, NHY, NH);
  pack_B_frag<1><<<4096, 256, 0, stream>>>(why, wyy, BYP, NHY, NY);

  gemm_iu8<0><<<dim3(NHY / 128, N_ROWS / 64, 1), 128, 0, stream>>>(
      X8, NX, WXP, nullptr, NX, H0, nullptr, nullptr, 0);
  reduce_max_i16<<<1024, 256, 0, stream>>>(H0, N_ROWS * NHY, TST);

  for (int it = 0; it < MAX_ITERS; ++it) {
    const signed char* cur = (it & 1) ? HY1 : HY0;
    signed char*       nxt = (it & 1) ? HY0 : HY1;
    gemm_iu8<1><<<dim3(NH / 128, N_ROWS / 64, 2), 128, 0, stream>>>(
        cur, NHY, BHP, BYP, NHY, H0, nxt, TST, it);
  }

  finalize<<<N_ROWS, 256, 0, stream>>>(HY0, H0, why, out);
}
